// CausalSelfAttention_6588479832230
// MI455X (gfx1250) — matrix operand order repaired and checked by reading, not compiled, not run on hardware
//
#include <hip/hip_runtime.h>
#include <hip/hip_bf16.h>
#include <math.h>

// ---------------- types ----------------
typedef __bf16 bf16;
typedef __attribute__((ext_vector_type(16))) bf16  bf16x16;
typedef __attribute__((ext_vector_type(8)))  bf16  bf16x8;
typedef __attribute__((ext_vector_type(8)))  float floatx8;

#define WMMA_BF16(a, b, c) \
  __builtin_amdgcn_wmma_f32_16x16x32_bf16(false, (a), false, (b), (short)0, (c), false, false)

// ---------------- problem dims ----------------
constexpr int Bn = 2, Tn = 2048, Cn = 1024, Hn = 16, Dn = 64;
constexpr int Mrows = Bn * Tn;     // 4096
constexpr int N3 = 3 * Cn;         // 3072
constexpr int BH = Bn * Hn;        // 32

__device__ inline floatx8 zero8() {
  floatx8 z;
#pragma unroll
  for (int i = 0; i < 8; ++i) z[i] = 0.0f;
  return z;
}

// A-fragment (16x32 bf16, MxK), tile stored row-major with leading dim ldk.
// lane = m + 16*h holds K = [8h, 8h+8) and [16+8h, 16+8h+8).
__device__ inline bf16x16 load_frag_a(const bf16* p, int ldk) {
  int lane = threadIdx.x & 31;
  int m = lane & 15, h = lane >> 4;
  const bf16* q = p + (size_t)m * ldk + 8 * h;
  bf16x8 lo = *(const bf16x8*)q;
  bf16x8 hi = *(const bf16x8*)(q + 16);
  bf16x16 r;
#pragma unroll
  for (int i = 0; i < 8; ++i) { r[i] = lo[i]; r[i + 8] = hi[i]; }
  return r;
}

// B-fragment (32x16 bf16, KxN). p points to B^T stored row-major [16][ldk]
// (row n = column n of B). lane = n + 16*h holds K = [8h, 8h+8) and [16+8h, 16+8h+8).
__device__ inline bf16x16 load_frag_b(const bf16* p, int ldk) {
  int lane = threadIdx.x & 31;
  int n = lane & 15, h = lane >> 4;
  const bf16* q = p + (size_t)n * ldk + 8 * h;
  bf16x8 lo = *(const bf16x8*)q;
  bf16x8 hi = *(const bf16x8*)(q + 16);
  bf16x16 r;
#pragma unroll
  for (int i = 0; i < 8; ++i) { r[i] = lo[i]; r[i + 8] = hi[i]; }
  return r;
}

// Per-lane prefetch of a fragment's cachelines (emits global_prefetch_b8).
__device__ inline void prefetch_frag(const bf16* p, int ldk) {
  int lane = threadIdx.x & 31;
  const bf16* q = p + (size_t)(lane & 15) * ldk + 16 * (lane >> 4);
  __builtin_prefetch((const void*)q, 0, 1);
}

// ---------------- conversion kernels ----------------
__global__ void cvt_f32_bf16(const float* __restrict__ in, bf16* __restrict__ out, int n) {
  int stride = gridDim.x * blockDim.x;
  for (int i = blockIdx.x * blockDim.x + threadIdx.x; i < n; i += stride)
    out[i] = (bf16)in[i];
}

// in: [rows][cols] f32 -> out: [cols][rows] bf16 (transposed)
__global__ void transpose_f32_bf16(const float* __restrict__ in, bf16* __restrict__ out,
                                   int rows, int cols) {
  int n = rows * cols;
  int stride = gridDim.x * blockDim.x;
  for (int i = blockIdx.x * blockDim.x + threadIdx.x; i < n; i += stride) {
    int c = i / rows, r = i - c * rows;
    out[(size_t)c * rows + r] = (bf16)in[(size_t)r * cols + c];
  }
}

// ---------------- shared GEMM core: 16(M) x 64(N), K = Cn, ping-pong pipelined ----
// Computes acc[4] (one 16x16 per N-chunk) for A row-major [16][Cn] at Abase and
// B^T rows at Bbase[c]. Unrolled by 2 so fragment buffers ping-pong (no reg copies).
__device__ inline void gemm_core(const bf16* Abase, const bf16* const Bbase[4],
                                 floatx8 acc[4]) {
  bf16x16 aA = load_frag_a(Abase, Cn);
  bf16x16 bA0 = load_frag_b(Bbase[0], Cn);
  bf16x16 bA1 = load_frag_b(Bbase[1], Cn);
  bf16x16 bA2 = load_frag_b(Bbase[2], Cn);
  bf16x16 bA3 = load_frag_b(Bbase[3], Cn);

  for (int k0 = 0; k0 < Cn - 64; k0 += 64) {
    // load ping buffer (k0+32) while computing with pong (k0)
    bf16x16 aB  = load_frag_a(Abase + k0 + 32, Cn);
    bf16x16 bB0 = load_frag_b(Bbase[0] + k0 + 32, Cn);
    bf16x16 bB1 = load_frag_b(Bbase[1] + k0 + 32, Cn);
    bf16x16 bB2 = load_frag_b(Bbase[2] + k0 + 32, Cn);
    bf16x16 bB3 = load_frag_b(Bbase[3] + k0 + 32, Cn);
    prefetch_frag(Abase + k0 + 96, Cn);
    prefetch_frag(Bbase[0] + k0 + 96, Cn);
    prefetch_frag(Bbase[2] + k0 + 96, Cn);
    acc[0] = WMMA_BF16(aA, bA0, acc[0]);
    acc[1] = WMMA_BF16(aA, bA1, acc[1]);
    acc[2] = WMMA_BF16(aA, bA2, acc[2]);
    acc[3] = WMMA_BF16(aA, bA3, acc[3]);
    // load pong buffer (k0+64) while computing with ping (k0+32)
    aA  = load_frag_a(Abase + k0 + 64, Cn);
    bA0 = load_frag_b(Bbase[0] + k0 + 64, Cn);
    bA1 = load_frag_b(Bbase[1] + k0 + 64, Cn);
    bA2 = load_frag_b(Bbase[2] + k0 + 64, Cn);
    bA3 = load_frag_b(Bbase[3] + k0 + 64, Cn);
    acc[0] = WMMA_BF16(aB, bB0, acc[0]);
    acc[1] = WMMA_BF16(aB, bB1, acc[1]);
    acc[2] = WMMA_BF16(aB, bB2, acc[2]);
    acc[3] = WMMA_BF16(aB, bB3, acc[3]);
  }
  // epilogue: pong holds k = Cn-64; ping covers k = Cn-32
  {
    bf16x16 aB  = load_frag_a(Abase + Cn - 32, Cn);
    bf16x16 bB0 = load_frag_b(Bbase[0] + Cn - 32, Cn);
    bf16x16 bB1 = load_frag_b(Bbase[1] + Cn - 32, Cn);
    bf16x16 bB2 = load_frag_b(Bbase[2] + Cn - 32, Cn);
    bf16x16 bB3 = load_frag_b(Bbase[3] + Cn - 32, Cn);
    acc[0] = WMMA_BF16(aA, bA0, acc[0]);
    acc[1] = WMMA_BF16(aA, bA1, acc[1]);
    acc[2] = WMMA_BF16(aA, bA2, acc[2]);
    acc[3] = WMMA_BF16(aA, bA3, acc[3]);
    acc[0] = WMMA_BF16(aB, bB0, acc[0]);
    acc[1] = WMMA_BF16(aB, bB1, acc[1]);
    acc[2] = WMMA_BF16(aB, bB2, acc[2]);
    acc[3] = WMMA_BF16(aB, bB3, acc[3]);
  }
}

// ---------------- QKV projection ----------------
// qkv = x @ W_qkv + b ; Q scaled by 1/sqrt(D); Q,K -> [bh][t][d], V -> [bh][d][t]
__global__ void __launch_bounds__(128) gemm_qkv(const bf16* __restrict__ xb,
                                                const bf16* __restrict__ wT,
                                                const float* __restrict__ bias,
                                                bf16* __restrict__ Qb,
                                                bf16* __restrict__ Kb,
                                                bf16* __restrict__ Vt) {
  const int lane = threadIdx.x & 31;
  const int wid = threadIdx.x >> 5;
  int wg = blockIdx.x * 4 + wid;
  int mt = wg / (N3 / 64), nt = wg % (N3 / 64);
  int m0 = mt * 16, n0 = nt * 64;

  const bf16* Abase = xb + (size_t)m0 * Cn;
  const bf16* Bbase[4];
#pragma unroll
  for (int c = 0; c < 4; ++c) Bbase[c] = wT + (size_t)(n0 + 16 * c) * Cn;

  floatx8 acc[4];
#pragma unroll
  for (int c = 0; c < 4; ++c) acc[c] = zero8();

  gemm_core(Abase, Bbase, acc);

  const int nidx = lane & 15, half = lane >> 4;
#pragma unroll
  for (int c = 0; c < 4; ++c) {
    int n = n0 + c * 16 + nidx;
    float bv = bias[n];
    int r = n >> 10;            // 0=Q 1=K 2=V (uniform per wave)
    int cc = n & (Cn - 1);
    int h = cc >> 6, d = cc & 63;
#pragma unroll
    for (int j = 0; j < 8; ++j) {
      int m = m0 + j + 8 * half;
      int b = m >> 11, t = m & (Tn - 1);
      int bh = b * Hn + h;
      float v = acc[c][j] + bv;
      if (r == 0)       Qb[((size_t)bh * Tn + t) * Dn + d] = (bf16)(v * 0.125f);
      else if (r == 1)  Kb[((size_t)bh * Tn + t) * Dn + d] = (bf16)v;
      else              Vt[((size_t)bh * Dn + d) * Tn + t] = (bf16)v;
    }
  }
}

// ---------------- flash attention (causal, online softmax) ----------------
__global__ void __launch_bounds__(128) attn_fwd(const bf16* __restrict__ Qb,
                                                const bf16* __restrict__ Kb,
                                                const bf16* __restrict__ Vt,
                                                bf16* __restrict__ Ob) {
  __shared__ __align__(16) bf16 pshare[4][16 * 32];
  const int lane = threadIdx.x & 31;
  const int wid = threadIdx.x >> 5;
  bf16* pl = &pshare[wid][0];

  int wg = blockIdx.x * 4 + wid;          // 4096 waves total
  int bh = wg >> 7;                       // / (T/16)
  int t0 = (wg & 127) << 4;
  const int nidx = lane & 15, half = lane >> 4;

  const bf16* Qrow = Qb + ((size_t)bh * Tn + t0) * Dn;
  bf16x16 qa0 = load_frag_a(Qrow, Dn);         // d 0..31
  bf16x16 qa1 = load_frag_a(Qrow + 32, Dn);    // d 32..63

  floatx8 o0 = zero8(), o1 = zero8(), o2 = zero8(), o3 = zero8();
  float mrow[8], lrow[8];
#pragma unroll
  for (int j = 0; j < 8; ++j) { mrow[j] = -__builtin_inff(); lrow[j] = 0.0f; }

  const bf16* Vbh = Vt + (size_t)bh * Dn * Tn;

  for (int s0 = 0; s0 <= t0 + 15; s0 += 32) {
    const bf16* Krow = Kb + ((size_t)bh * Tn + s0) * Dn;

    // S = Q K^T for 32 key columns (4 WMMAs)
    bf16x16 kb00 = load_frag_b(Krow, Dn);
    bf16x16 kb01 = load_frag_b(Krow + 32, Dn);
    bf16x16 kb10 = load_frag_b(Krow + 16 * Dn, Dn);
    bf16x16 kb11 = load_frag_b(Krow + 16 * Dn + 32, Dn);
    floatx8 S0 = zero8(), S1 = zero8();
    S0 = WMMA_BF16(qa0, kb00, S0);
    S0 = WMMA_BF16(qa1, kb01, S0);
    S1 = WMMA_BF16(qa0, kb10, S1);
    S1 = WMMA_BF16(qa1, kb11, S1);

    // issue V fragment loads now; softmax VALU below hides their latency
    bf16x16 vb0 = load_frag_b(Vbh + (size_t)0  * Tn + s0, Tn);
    bf16x16 vb1 = load_frag_b(Vbh + (size_t)16 * Tn + s0, Tn);
    bf16x16 vb2 = load_frag_b(Vbh + (size_t)32 * Tn + s0, Tn);
    bf16x16 vb3 = load_frag_b(Vbh + (size_t)48 * Tn + s0, Tn);

    // prefetch next K tile
    if (s0 + 32 < Tn) {
      prefetch_frag(Krow + 32 * Dn, Dn);
      prefetch_frag(Krow + 48 * Dn, Dn);
    }

#pragma unroll
    for (int j = 0; j < 8; ++j) {
      int trow = t0 + j + 8 * half;
      if (s0 + nidx > trow)      S0[j] = -__builtin_inff();
      if (s0 + 16 + nidx > trow) S1[j] = -__builtin_inff();
      float cand = fmaxf(S0[j], S1[j]);
      cand = fmaxf(cand, __shfl_xor(cand, 1, 32));
      cand = fmaxf(cand, __shfl_xor(cand, 2, 32));
      cand = fmaxf(cand, __shfl_xor(cand, 4, 32));
      cand = fmaxf(cand, __shfl_xor(cand, 8, 32));
      float mnew = fmaxf(mrow[j], cand);
      float p0 = __expf(S0[j] - mnew);
      float p1 = __expf(S1[j] - mnew);
      float rs = p0 + p1;
      rs += __shfl_xor(rs, 1, 32);
      rs += __shfl_xor(rs, 2, 32);
      rs += __shfl_xor(rs, 4, 32);
      rs += __shfl_xor(rs, 8, 32);
      float scale = __expf(mrow[j] - mnew);
      lrow[j] = lrow[j] * scale + rs;
      mrow[j] = mnew;
      o0[j] *= scale; o1[j] *= scale; o2[j] *= scale; o3[j] *= scale;
      int row = j + 8 * half;
      pl[row * 32 + nidx]      = (bf16)p0;
      pl[row * 32 + 16 + nidx] = (bf16)p1;
    }
    asm volatile("s_wait_dscnt 0" ::: "memory");
    __builtin_amdgcn_wave_barrier();

    bf16x16 pa = load_frag_a(pl, 32);           // P as 16x32 A-fragment
    o0 = WMMA_BF16(pa, vb0, o0);
    o1 = WMMA_BF16(pa, vb1, o1);
    o2 = WMMA_BF16(pa, vb2, o2);
    o3 = WMMA_BF16(pa, vb3, o3);

    asm volatile("s_wait_dscnt 0" ::: "memory");
    __builtin_amdgcn_wave_barrier();
  }

  int b = bh >> 4, h = bh & 15;
#pragma unroll
  for (int j = 0; j < 8; ++j) {
    float inv = 1.0f / lrow[j];
    int t = t0 + j + 8 * half;
    size_t ro = ((size_t)b * Tn + t) * Cn + h * Dn;
    Ob[ro + 0  + nidx] = (bf16)(o0[j] * inv);
    Ob[ro + 16 + nidx] = (bf16)(o1[j] * inv);
    Ob[ro + 32 + nidx] = (bf16)(o2[j] * inv);
    Ob[ro + 48 + nidx] = (bf16)(o3[j] * inv);
  }
}

// ---------------- output projection ----------------
__global__ void __launch_bounds__(128) gemm_out(const bf16* __restrict__ Ob,
                                                const bf16* __restrict__ wT,
                                                const float* __restrict__ bias,
                                                float* __restrict__ out) {
  const int lane = threadIdx.x & 31;
  const int wid = threadIdx.x >> 5;
  int wg = blockIdx.x * 4 + wid;
  int mt = wg >> 4, nt = wg & 15;
  int m0 = mt * 16, n0 = nt * 64;

  const bf16* Abase = Ob + (size_t)m0 * Cn;
  const bf16* Bbase[4];
#pragma unroll
  for (int c = 0; c < 4; ++c) Bbase[c] = wT + (size_t)(n0 + 16 * c) * Cn;

  floatx8 acc[4];
#pragma unroll
  for (int c = 0; c < 4; ++c) acc[c] = zero8();

  gemm_core(Abase, Bbase, acc);

  const int nidx = lane & 15, half = lane >> 4;
#pragma unroll
  for (int c = 0; c < 4; ++c) {
    int n = n0 + c * 16 + nidx;
    float bv = bias[n];
#pragma unroll
    for (int j = 0; j < 8; ++j) {
      int m = m0 + j + 8 * half;
      out[(size_t)m * Cn + n] = acc[c][j] + bv;
    }
  }
}

// ---------------- host ----------------
extern "C" void kernel_launch(void* const* d_in, const int* in_sizes, int n_in,
                              void* d_out, int out_size, void* d_ws, size_t ws_size,
                              hipStream_t stream) {
  (void)in_sizes; (void)n_in; (void)out_size; (void)ws_size;
  const float* x    = (const float*)d_in[0];
  const float* Wqkv = (const float*)d_in[1];
  const float* bqkv = (const float*)d_in[2];
  const float* Wout = (const float*)d_in[3];
  const float* bout = (const float*)d_in[4];
  float* out = (float*)d_out;

  char* ws = (char*)d_ws;
  bf16* xb    = (bf16*)ws;                          ws += (size_t)Mrows * Cn * 2;   // 8 MB
  bf16* wqkvT = (bf16*)ws;                          ws += (size_t)N3 * Cn * 2;      // 6 MB
  bf16* woutT = (bf16*)ws;                          ws += (size_t)Cn * Cn * 2;      // 2 MB
  bf16* Qb    = (bf16*)ws;                          ws += (size_t)BH * Tn * Dn * 2; // 8 MB
  bf16* Kb    = (bf16*)ws;                          ws += (size_t)BH * Tn * Dn * 2; // 8 MB
  bf16* Vt    = (bf16*)ws;                          ws += (size_t)BH * Dn * Tn * 2; // 8 MB
  bf16* Ob    = (bf16*)ws;                          ws += (size_t)Mrows * Cn * 2;   // 8 MB

  cvt_f32_bf16<<<2048, 256, 0, stream>>>(x, xb, Mrows * Cn);
  transpose_f32_bf16<<<1536, 256, 0, stream>>>(Wqkv, wqkvT, Cn, N3);
  transpose_f32_bf16<<<512, 256, 0, stream>>>(Wout, woutT, Cn, Cn);
  gemm_qkv<<<(Mrows / 16) * (N3 / 64) / 4, 128, 0, stream>>>(xb, wqkvT, bqkv, Qb, Kb, Vt);
  attn_fwd<<<BH * (Tn / 16) / 4, 128, 0, stream>>>(Qb, Kb, Vt, Ob);
  gemm_out<<<(Mrows / 16) * (Cn / 64) / 4, 128, 0, stream>>>(Ob, woutT, bout, out);
}
